// ChestXrayGNN_81896436400236
// MI455X (gfx1250) — matrix operand order repaired and checked by reading, not compiled, not run on hardware
//
#include <hip/hip_runtime.h>
#include <stdint.h>

#define HEADS 4
#define NNODE 257
#define NPAD  288   // 9 * 32 (GEMM M-tile = 32)
#define NB    32
#define C_IN  2048
#define D1V   256
#define D2V   128
#define N1    1024  // HEADS*D1
#define N2    512   // HEADS*D2

typedef __attribute__((ext_vector_type(16))) __bf16 v16bf_t;
typedef __attribute__((ext_vector_type(8)))  float  v8f_t;

union FragU { v16bf_t v; uint4 u[2]; };

__device__ __forceinline__ unsigned short f2bf(float f) {
  unsigned u = __float_as_uint(f);
  u += 0x7FFFu + ((u >> 16) & 1u);          // round-to-nearest-even
  return (unsigned short)(u >> 16);
}
__device__ __forceinline__ unsigned ordenc(float f) {
  unsigned u = __float_as_uint(f);
  return (u & 0x80000000u) ? ~u : (u | 0x80000000u);
}
__device__ __forceinline__ float orddec(unsigned u) {
  return __uint_as_float((u & 0x80000000u) ? (u ^ 0x80000000u) : ~u);
}
__device__ __forceinline__ float lrelu02(float v) { return v > 0.f ? v : 0.2f * v; }
__device__ __forceinline__ float sigm(float x) { return 1.f / (1.f + expf(-x)); }

// ---------------- utility fills ----------------
__global__ void k_fill_u32(unsigned* p, unsigned v, long long n) {
  long long t = (long long)blockIdx.x * blockDim.x + threadIdx.x;
  if (t < n) p[t] = v;
}

// ---------------- stage X1 (transpose fmap, fp32->bf16) ----------------
__global__ void k_prep_x1(const float* __restrict__ fmap, unsigned short* __restrict__ X1) {
  __shared__ float tile[16][17];
  int b = blockIdx.z;
  int c0 = blockIdx.x * 16, n0 = blockIdx.y * 16;
  int i = threadIdx.x >> 4, j = threadIdx.x & 15;
  tile[i][j] = fmap[((size_t)b * C_IN + c0 + i) * 256 + n0 + j];
  __syncthreads();
  // node row n0+i, channel c0+j  <=  fmap[b, c0+j, n0+i]
  X1[((size_t)b * NPAD + n0 + i) * C_IN + c0 + j] = f2bf(tile[j][i]);
}

// ---------------- stage W^T (fp32 [K,N] -> bf16 [N,K]) ----------------
__global__ void k_transpose_bf16(const float* __restrict__ W, unsigned short* __restrict__ WT,
                                 int K, int N) {
  long long t = (long long)blockIdx.x * blockDim.x + threadIdx.x;
  if (t >= (long long)K * N) return;
  int k = (int)(t % K);
  int o = (int)(t / K);
  WT[(size_t)o * K + k] = f2bf(W[(size_t)k * N + o]);
}

// ---------------- bf16 WMMA GEMM: C[b] = A[b] (M x K) * BT^T (K x N) ----------------
// A row-major bf16 (padded rows), BT = B^T row-major bf16.
// One wave computes a 32x64 tile: 2 A-frags x 4 B-frags -> 8 WMMAs per k-step.
__global__ void k_gemm_bf16(const unsigned short* __restrict__ A,
                            const unsigned short* __restrict__ BT,
                            float* __restrict__ Cc,
                            int Mvalid, int N, int K,
                            long long strideA, long long strideC) {
  int b = blockIdx.z, mt = blockIdx.x, nt = blockIdx.y;
  int lane = threadIdx.x & 31;
  int hf = lane >> 4, r = lane & 15;
  const unsigned short* Ar0 = A + (size_t)b * strideA + (size_t)(mt * 32 + r) * K;
  const unsigned short* Ar1 = Ar0 + (size_t)16 * K;
  const unsigned short* Br  = BT + (size_t)(nt * 64 + r) * K;
  v8f_t acc[2][4];
#pragma unroll
  for (int mi = 0; mi < 2; ++mi)
#pragma unroll
    for (int j = 0; j < 4; ++j) acc[mi][j] = (v8f_t){};

  for (int k = 0; k < K; k += 32) {
    FragU a0, a1, bf[4];
    // A 16x32 bf16 layout: lane half hf holds K = hf*8+{0..7} and 16+hf*8+{0..7}
    a0.u[0] = *(const uint4*)(Ar0 + k + hf * 8);
    a0.u[1] = *(const uint4*)(Ar0 + k + 16 + hf * 8);
    a1.u[0] = *(const uint4*)(Ar1 + k + hf * 8);
    a1.u[1] = *(const uint4*)(Ar1 + k + 16 + hf * 8);
    // B 32x16 bf16 layout: column r, K = hf*8+{0..7} and 16+hf*8+{0..7} (W^T row)
#pragma unroll
    for (int j = 0; j < 4; ++j) {
      const unsigned short* Bj = Br + (size_t)(16 * j) * K;
      bf[j].u[0] = *(const uint4*)(Bj + k + hf * 8);
      bf[j].u[1] = *(const uint4*)(Bj + k + 16 + hf * 8);
    }
#pragma unroll
    for (int j = 0; j < 4; ++j) {
      acc[0][j] = __builtin_amdgcn_wmma_f32_16x16x32_bf16(false, a0.v, false, bf[j].v,
                                                          (short)0, acc[0][j], false, false);
      acc[1][j] = __builtin_amdgcn_wmma_f32_16x16x32_bf16(false, a1.v, false, bf[j].v,
                                                          (short)0, acc[1][j], false, false);
    }
  }
#pragma unroll
  for (int mi = 0; mi < 2; ++mi) {
    int rowbase = mt * 32 + mi * 16 + 8 * hf;
#pragma unroll
    for (int j = 0; j < 4; ++j) {
#pragma unroll
      for (int rr = 0; rr < 8; ++rr) {
        int row = rowbase + rr;
        if (row < Mvalid)
          Cc[(size_t)b * strideC + (size_t)row * N + nt * 64 + j * 16 + r] = acc[mi][j][rr];
      }
    }
  }
}

// ---------------- per-(node,head) attention coefficients ----------------
__global__ void k_es_ed(const float* __restrict__ Hm, const float* __restrict__ as_,
                        const float* __restrict__ ad_, float* __restrict__ es,
                        float* __restrict__ ed, int D) {
  int t = blockIdx.x * blockDim.x + threadIdx.x;
  if (t >= NB * NNODE * HEADS) return;
  int h = t & 3;                       // t == ((b*257+n)*4 + h)
  const float* hp = Hm + (size_t)t * D;
  float s = 0.f, q = 0.f;
  for (int d = 0; d < D; ++d) {
    float v = hp[d];
    s += v * as_[h * D + d];
    q += v * ad_[h * D + d];
  }
  es[t] = s; ed[t] = q;
}

__global__ void k_edge_max(const float* __restrict__ es, const float* __restrict__ ed,
                           const int* __restrict__ src, const int* __restrict__ dst,
                           unsigned* __restrict__ mx, int E) {
  long long t = (long long)blockIdx.x * blockDim.x + threadIdx.x;
  if (t >= (long long)NB * E * HEADS) return;
  int h = (int)(t & 3);
  int e = (int)((t >> 2) % E);
  int b = (int)(t / ((long long)E * HEADS));
  int sn = src[e], dn = dst[e];
  float v = lrelu02(es[((size_t)b * NNODE + sn) * HEADS + h] +
                    ed[((size_t)b * NNODE + dn) * HEADS + h]);
  atomicMax(&mx[((size_t)b * NNODE + dn) * HEADS + h], ordenc(v));
}

__global__ void k_edge_alpha(const float* __restrict__ es, const float* __restrict__ ed,
                             const int* __restrict__ src, const int* __restrict__ dst,
                             const unsigned* __restrict__ mx, float* __restrict__ den,
                             float* __restrict__ alpha, int E) {
  long long t = (long long)blockIdx.x * blockDim.x + threadIdx.x;
  if (t >= (long long)NB * E * HEADS) return;
  int h = (int)(t & 3);
  int e = (int)((t >> 2) % E);
  int b = (int)(t / ((long long)E * HEADS));
  int sn = src[e], dn = dst[e];
  size_t dIdx = ((size_t)b * NNODE + dn) * HEADS + h;
  float v = lrelu02(es[((size_t)b * NNODE + sn) * HEADS + h] + ed[dIdx]);
  float a = expf(v - orddec(mx[dIdx]));
  alpha[((size_t)b * E + e) * HEADS + h] = a;
  atomicAdd(&den[dIdx], a);
}

__global__ void k_edge_agg(const float* __restrict__ Hm, const int* __restrict__ src,
                           const int* __restrict__ dst, const float* __restrict__ alpha,
                           const float* __restrict__ den, float* __restrict__ agg,
                           int E, int D) {
  long long t = (long long)blockIdx.x * blockDim.x + threadIdx.x;
  long long total = (long long)NB * E * HEADS * D;
  if (t >= total) return;
  int d = (int)(t % D);
  long long t2 = t / D;
  int h = (int)(t2 & 3);
  long long t3 = t2 >> 2;
  int e = (int)(t3 % E);
  int b = (int)(t3 / E);
  int sn = src[e], dn = dst[e];
  size_t dIdx = ((size_t)b * NNODE + dn) * HEADS + h;
  float coef = alpha[((size_t)b * E + e) * HEADS + h] / (den[dIdx] + 1e-16f);
  atomicAdd(&agg[dIdx * D + d],
            coef * Hm[(((size_t)b * NNODE + sn) * HEADS + h) * D + d]);
}

// mean over heads + bias; write bf16 padded rows for next GEMM
__global__ void k_finalize1(const float* __restrict__ agg, const float* __restrict__ bias,
                            unsigned short* __restrict__ Xout) {
  int t = blockIdx.x * blockDim.x + threadIdx.x;
  if (t >= NB * NNODE * D1V) return;
  int d = t % D1V;
  int n = (t / D1V) % NNODE;
  int b = t / (D1V * NNODE);
  size_t base = ((size_t)b * NNODE + n) * HEADS;
  float v = 0.25f * (agg[(base + 0) * D1V + d] + agg[(base + 1) * D1V + d] +
                     agg[(base + 2) * D1V + d] + agg[(base + 3) * D1V + d]) + bias[d];
  Xout[((size_t)b * NPAD + n) * D1V + d] = f2bf(v);
}

__global__ void k_finalize2(const float* __restrict__ agg, const float* __restrict__ bias,
                            float* __restrict__ X3) {
  int t = blockIdx.x * blockDim.x + threadIdx.x;
  if (t >= NB * NNODE * D2V) return;
  int d = t % D2V;
  int n = (t / D2V) % NNODE;
  int b = t / (D2V * NNODE);
  size_t base = ((size_t)b * NNODE + n) * HEADS;
  float v = 0.25f * (agg[(base + 0) * D2V + d] + agg[(base + 1) * D2V + d] +
                     agg[(base + 2) * D2V + d] + agg[(base + 3) * D2V + d]) + bias[d];
  X3[((size_t)b * NNODE + n) * D2V + d] = v;
}

// ---------------- pooling + classifier, one workgroup per graph ----------------
__global__ void k_pool(const float* __restrict__ x3, const float* __restrict__ Wp,
                       const float* __restrict__ bp, const float* __restrict__ Wc1,
                       const float* __restrict__ bc1, const float* __restrict__ Wc2,
                       const float* __restrict__ bc2, float* __restrict__ preds,
                       float* __restrict__ atts) {
  __shared__ float wbuf[NNODE];
  __shared__ float gbuf[D2V];
  __shared__ float h1buf[64];
  int b = blockIdx.x, t = threadIdx.x;
  const float* X = x3 + (size_t)b * NNODE * D2V;
  for (int n = t; n < NNODE; n += 256) {
    float s = bp[0];
    for (int d = 0; d < D2V; ++d) s += X[n * D2V + d] * Wp[d];
    float w = sigm(s);
    wbuf[n] = w;
    atts[(size_t)b * NNODE + n] = w;
  }
  __syncthreads();
  if (t < D2V) {
    float s = 0.f;
    for (int n = 0; n < NNODE; ++n) s += X[n * D2V + t] * wbuf[n];
    gbuf[t] = s / (float)NNODE;
  }
  __syncthreads();
  if (t < 64) {
    float s = bc1[t];
    for (int d = 0; d < D2V; ++d) s += gbuf[d] * Wc1[d * 64 + t];
    h1buf[t] = fmaxf(s, 0.f);
  }
  __syncthreads();
  if (t == 0) {
    float s = bc2[0];
    for (int j = 0; j < 64; ++j) s += h1buf[j] * Wc2[j];
    preds[b] = sigm(s);
  }
}

static inline unsigned cdiv(long long n, int b) { return (unsigned)((n + b - 1) / b); }

extern "C" void kernel_launch(void* const* d_in, const int* in_sizes, int n_in,
                              void* d_out, int out_size, void* d_ws, size_t ws_size,
                              hipStream_t stream) {
  (void)n_in; (void)out_size; (void)ws_size;
  const float* fmap  = (const float*)d_in[0];
  const float* W1    = (const float*)d_in[1];
  const float* as1   = (const float*)d_in[2];
  const float* ad1   = (const float*)d_in[3];
  const float* b1    = (const float*)d_in[4];
  const float* W2    = (const float*)d_in[5];
  const float* as2   = (const float*)d_in[6];
  const float* ad2   = (const float*)d_in[7];
  const float* b2    = (const float*)d_in[8];
  const float* Wp    = (const float*)d_in[9];
  const float* bp    = (const float*)d_in[10];
  const float* Wc1   = (const float*)d_in[11];
  const float* bc1   = (const float*)d_in[12];
  const float* Wc2   = (const float*)d_in[13];
  const float* bc2   = (const float*)d_in[14];
  const int*   src   = (const int*)d_in[15];
  const int*   dst   = (const int*)d_in[16];
  const int E = in_sizes[15];

  float* out   = (float*)d_out;
  float* preds = out;        // [32]
  float* atts  = out + NB;   // [32*257]

  // ---- workspace carve-out ----
  char* wp = (char*)d_ws;
  size_t off = 0;
  auto alloc = [&](size_t bytes) -> void* {
    void* r = (void*)(wp + off);
    off += (bytes + 255) & ~(size_t)255;
    return r;
  };
  unsigned short* W1T  = (unsigned short*)alloc((size_t)N1 * C_IN * 2);
  unsigned short* W2T  = (unsigned short*)alloc((size_t)N2 * D1V * 2);
  unsigned short* X1   = (unsigned short*)alloc((size_t)NB * NPAD * C_IN * 2);
  unsigned short* X2   = (unsigned short*)alloc((size_t)NB * NPAD * D1V * 2);
  float* H1   = (float*)alloc((size_t)NB * NNODE * N1 * 4);
  float* H2   = (float*)alloc((size_t)NB * NNODE * N2 * 4);
  float* AGG1 = (float*)alloc((size_t)NB * NNODE * N1 * 4);
  float* AGG2 = (float*)alloc((size_t)NB * NNODE * N2 * 4);
  float* ES   = (float*)alloc((size_t)NB * NNODE * HEADS * 4);
  float* ED   = (float*)alloc((size_t)NB * NNODE * HEADS * 4);
  unsigned* MX = (unsigned*)alloc((size_t)NB * NNODE * HEADS * 4);
  float* DEN  = (float*)alloc((size_t)NB * NNODE * HEADS * 4);
  float* ALPHA = (float*)alloc((size_t)NB * E * HEADS * 4);
  float* X3   = (float*)alloc((size_t)NB * NNODE * D2V * 4);

  const long long nX1u = (long long)NB * NPAD * C_IN / 2;  // u32 count
  const long long nX2u = (long long)NB * NPAD * D1V / 2;
  const long long nAg1 = (long long)NB * NNODE * N1;
  const long long nAg2 = (long long)NB * NNODE * N2;
  const long long nNH  = (long long)NB * NNODE * HEADS;
  const long long nEH  = (long long)NB * E * HEADS;

  // ---- stage inputs ----
  k_fill_u32<<<cdiv(nX1u, 256), 256, 0, stream>>>((unsigned*)X1, 0u, nX1u);
  k_fill_u32<<<cdiv(nX2u, 256), 256, 0, stream>>>((unsigned*)X2, 0u, nX2u);
  k_prep_x1<<<dim3(C_IN / 16, 256 / 16, NB), 256, 0, stream>>>(fmap, X1);
  k_transpose_bf16<<<cdiv((long long)C_IN * N1, 256), 256, 0, stream>>>(W1, W1T, C_IN, N1);
  k_transpose_bf16<<<cdiv((long long)D1V * N2, 256), 256, 0, stream>>>(W2, W2T, D1V, N2);

  // ---- layer 1: GEMM + GAT attention ----
  k_gemm_bf16<<<dim3(NPAD / 32, N1 / 64, NB), 32, 0, stream>>>(
      X1, W1T, H1, NNODE, N1, C_IN, (long long)NPAD * C_IN, (long long)NNODE * N1);
  k_es_ed<<<cdiv(nNH, 256), 256, 0, stream>>>(H1, as1, ad1, ES, ED, D1V);
  k_fill_u32<<<cdiv(nNH, 256), 256, 0, stream>>>(MX, 0u, nNH);
  k_fill_u32<<<cdiv(nNH, 256), 256, 0, stream>>>((unsigned*)DEN, 0u, nNH);
  k_fill_u32<<<cdiv(nAg1, 256), 256, 0, stream>>>((unsigned*)AGG1, 0u, nAg1);
  k_edge_max<<<cdiv(nEH, 256), 256, 0, stream>>>(ES, ED, src, dst, MX, E);
  k_edge_alpha<<<cdiv(nEH, 256), 256, 0, stream>>>(ES, ED, src, dst, MX, DEN, ALPHA, E);
  k_edge_agg<<<cdiv(nEH * D1V, 256), 256, 0, stream>>>(H1, src, dst, ALPHA, DEN, AGG1, E, D1V);
  k_finalize1<<<cdiv((long long)NB * NNODE * D1V, 256), 256, 0, stream>>>(AGG1, b1, X2);

  // ---- layer 2: GEMM + GAT attention ----
  k_gemm_bf16<<<dim3(NPAD / 32, N2 / 64, NB), 32, 0, stream>>>(
      X2, W2T, H2, NNODE, N2, D1V, (long long)NPAD * D1V, (long long)NNODE * N2);
  k_es_ed<<<cdiv(nNH, 256), 256, 0, stream>>>(H2, as2, ad2, ES, ED, D2V);
  k_fill_u32<<<cdiv(nNH, 256), 256, 0, stream>>>(MX, 0u, nNH);
  k_fill_u32<<<cdiv(nNH, 256), 256, 0, stream>>>((unsigned*)DEN, 0u, nNH);
  k_fill_u32<<<cdiv(nAg2, 256), 256, 0, stream>>>((unsigned*)AGG2, 0u, nAg2);
  k_edge_max<<<cdiv(nEH, 256), 256, 0, stream>>>(ES, ED, src, dst, MX, E);
  k_edge_alpha<<<cdiv(nEH, 256), 256, 0, stream>>>(ES, ED, src, dst, MX, DEN, ALPHA, E);
  k_edge_agg<<<cdiv(nEH * D2V, 256), 256, 0, stream>>>(H2, src, dst, ALPHA, DEN, AGG2, E, D2V);
  k_finalize2<<<cdiv((long long)NB * NNODE * D2V, 256), 256, 0, stream>>>(AGG2, b2, X3);

  // ---- pooling + classifier ----
  k_pool<<<NB, 256, 0, stream>>>(X3, Wp, bp, Wc1, bc1, Wc2, bc2, preds, atts);
}
